// HAGNN_Layer_3453153706263
// MI455X (gfx1250) — matrix operand order repaired and checked by reading, not compiled, not run on hardware
//
#include <hip/hip_runtime.h>
#include <hip/hip_bf16.h>
#include <math.h>

typedef __attribute__((ext_vector_type(16))) __bf16 v16bf;
typedef __attribute__((ext_vector_type(8)))  __bf16 v8bf;
typedef __attribute__((ext_vector_type(8)))  float  v8f;

#define N_NODES 10000
#define IN_F    512
#define OUT_F   64
#define ALPHA   0.2f
#define THRESH  0.01f
#define WHT_STRIDE 10240   // padded column count (multiple of 1024)
#define BITS_WORDS 320     // 320*32 = 10240 packed adjacency bits per row
#define NCH     10         // column chunks for aggregation pass
#define CHUNK   1024

// ---- workspace layout (bytes) ----
static constexpr size_t HB_OFF   = 0;                                   // h in bf16 [N][512]
static constexpr size_t HB_SZ    = (size_t)N_NODES * IN_F * 2;
static constexpr size_t WT_OFF   = HB_OFF + HB_SZ;                      // W^T bf16 [64][512]
static constexpr size_t WT_SZ    = (size_t)OUT_F * IN_F * 2;
static constexpr size_t WHT_OFF  = WT_OFF + WT_SZ;                      // Wh^T bf16 [64][10240]
static constexpr size_t WHT_SZ   = (size_t)OUT_F * WHT_STRIDE * 2;
static constexpr size_t ES_OFF   = WHT_OFF + WHT_SZ;                    // e_src f32 [N]
static constexpr size_t ED_OFF   = ES_OFF + (size_t)N_NODES * 4;        // e_dst f32 [N]
static constexpr size_t M_OFF    = ED_OFF + (size_t)N_NODES * 4;        // row max f32 [N]
static constexpr size_t IZ_OFF   = M_OFF  + (size_t)N_NODES * 4;        // 1/Z f32 [N]
static constexpr size_t BITS_OFF = IZ_OFF + (size_t)N_NODES * 4;        // adjacency bitmask
static constexpr size_t BITS_SZ  = (size_t)N_NODES * BITS_WORDS * 4;
static constexpr size_t PART_OFF = BITS_OFF + BITS_SZ;                  // partials [NCH][N][64] f32

// ---------- kernel 0a: h -> bf16 ----------
__global__ void k_cvt_h(const float* __restrict__ h, __bf16* __restrict__ hB) {
    size_t i = (size_t)blockIdx.x * blockDim.x + threadIdx.x;
    if (i < (size_t)N_NODES * IN_F) hB[i] = (__bf16)h[i];
}

// ---------- kernel 0b: W -> W^T bf16, and zero WhBT column padding ----------
__global__ void k_cvt_w(const float* __restrict__ W, __bf16* __restrict__ WT,
                        __bf16* __restrict__ WhBT) {
    int i = blockIdx.x * blockDim.x + threadIdx.x;
    if (i < IN_F * OUT_F) {
        int k = i / OUT_F, n = i % OUT_F;
        WT[(size_t)n * IN_F + k] = (__bf16)W[i];
    }
    const int PAD = WHT_STRIDE - N_NODES; // 240
    if (i < OUT_F * PAD) {
        int n = i / PAD, p = i % PAD;
        WhBT[(size_t)n * WHT_STRIDE + N_NODES + p] = (__bf16)0.f;
    }
}

// ---------- kernel 1: Wh = h@W via bf16 WMMA; also e_src/e_dst ----------
// block = 128 threads (4 waves); each wave computes one 16x16 tile of Wh.
__global__ __launch_bounds__(128) void k_wh(const __bf16* __restrict__ hB,
                                            const __bf16* __restrict__ WT,
                                            const float* __restrict__ a_src,
                                            const float* __restrict__ a_dst,
                                            __bf16* __restrict__ WhBT,
                                            float* __restrict__ es,
                                            float* __restrict__ ed) {
    __shared__ float ldsWh[16 * OUT_F];
    const int tid  = threadIdx.x;
    const int wid  = tid >> 5, lane = tid & 31;
    const int i0   = blockIdx.x * 16;
    const int n0   = wid * 16;
    const int lrow = lane & 15;
    const int hi   = lane >> 4;

    v8f acc = {};
    const __bf16* aRow = hB + (size_t)(i0 + lrow) * IN_F;
    const __bf16* bRow = WT + (size_t)(n0 + lrow) * IN_F;
    for (int kk = 0; kk < IN_F; kk += 32) {
        // A: lane half hi -> K = kk+hi*16+{0..15} (matches B)
        v8bf alo = *(const v8bf*)(aRow + kk + hi * 16);
        v8bf ahi = *(const v8bf*)(aRow + kk + hi * 16 + 8);
        v16bf a = __builtin_shufflevector(alo, ahi, 0,1,2,3,4,5,6,7,8,9,10,11,12,13,14,15);
        // B (contiguous k, as A): lanes 0-15 hold K=kk..kk+15, lanes 16-31 hold K=kk+16..kk+31
        v16bf b = *(const v16bf*)(bRow + kk + hi * 16);
        acc = __builtin_amdgcn_wmma_f32_16x16x32_bf16(false, a, false, b,
                                                      (short)0, acc, false, false);
    }
    // C layout: element v of lane -> (M = hi*8+v, N = lrow). Store Wh^T (bf16) + LDS (f32).
    v8bf cv;
#pragma unroll
    for (int v = 0; v < 8; ++v) {
        cv[v] = (__bf16)acc[v];
        ldsWh[(hi * 8 + v) * OUT_F + n0 + lrow] = acc[v];
    }
    *(v8bf*)(WhBT + (size_t)(n0 + lrow) * WHT_STRIDE + i0 + hi * 8) = cv;

    __syncthreads();
    if (tid < 16) {
        float s0 = 0.f, s1 = 0.f;
#pragma unroll
        for (int n = 0; n < OUT_F; ++n) {
            float w = ldsWh[tid * OUT_F + n];
            s0 += w * a_src[n];
            s1 += w * a_dst[n];
        }
        es[i0 + tid] = s0;
        ed[i0 + tid] = s1;
    }
}

// ---------- kernel 2: one pass over adj: pack bitmask + online (max, sumexp) ----------
__global__ __launch_bounds__(256) void k_stats(const int* __restrict__ adj,
                                               const float* __restrict__ ed,
                                               const float* __restrict__ es,
                                               unsigned int* __restrict__ bits,
                                               float* __restrict__ mrow,
                                               float* __restrict__ izrow) {
    __shared__ float lm[256], ls[256];
    const int i = blockIdx.x;
    const int tid = threadIdx.x;
    const int wid = tid >> 5, lane = tid & 31;
    const float esi = es[i];
    const int* arow = adj + (size_t)i * N_NODES;
    unsigned int* brow = bits + (size_t)i * BITS_WORDS;

    float m = -INFINITY, s = 0.f;
    for (int base = 0; base < BITS_WORDS * 32; base += 256) {
        int j = base + tid;
        bool pred = (j < N_NODES) && (arow[j < N_NODES ? j : 0] > 0);
        unsigned int bm = __builtin_amdgcn_ballot_w32(pred);
        if (lane == 0) brow[(base >> 5) + wid] = bm;
        if (pred) {
            float e = esi + ed[j];
            e = e > 0.f ? e : ALPHA * e;
            if (e > m) { s = s * __expf(m - e) + 1.f; m = e; }
            else       { s += __expf(e - m); }
        }
    }
    lm[tid] = m; ls[tid] = s;
    __syncthreads();
    for (int off = 128; off > 0; off >>= 1) {
        if (tid < off) {
            float m1 = lm[tid],       s1 = ls[tid];
            float m2 = lm[tid + off], s2 = ls[tid + off];
            float M, S;
            if (m2 == -INFINITY)      { M = m1; S = s1; }
            else if (m1 == -INFINITY) { M = m2; S = s2; }
            else { M = fmaxf(m1, m2); S = s1 * __expf(m1 - M) + s2 * __expf(m2 - M); }
            lm[tid] = M; ls[tid] = S;
        }
        __syncthreads();
    }
    if (tid == 0) {
        mrow[i]  = lm[0];
        izrow[i] = ls[0] > 0.f ? 1.f / ls[0] : 0.f;
    }
}

// ---------- kernel 3: h' partial = attn_chunk @ Wh via bf16 WMMA ----------
// one wave per (16-row tile, 1024-column chunk); 4 accumulators cover OUT_F=64.
__global__ __launch_bounds__(32) void k_aggr(const __bf16* __restrict__ WhBT,
                                             const float* __restrict__ ed,
                                             const float* __restrict__ es,
                                             const float* __restrict__ hub,
                                             const unsigned int* __restrict__ bits,
                                             const float* __restrict__ mrow,
                                             const float* __restrict__ izrow,
                                             float* __restrict__ part) {
    __shared__ float led[CHUNK];
    __shared__ float lth[CHUNK];
    const int lane = threadIdx.x;
    const int i0   = blockIdx.x * 16;
    const int ch   = blockIdx.y;
    const int jb0  = ch * CHUNK;

    for (int t = lane; t < CHUNK; t += 32) {
        int j = jb0 + t;
        bool ok = j < N_NODES;
        led[t] = ok ? ed[j] : 0.f;
        lth[t] = ok ? hub[j] * THRESH : 0.f;
    }
    __syncthreads();

    const int lrow = lane & 15;
    const int hi   = lane >> 4;
    const int row  = i0 + lrow;
    const float esi = es[row];
    const float mi  = mrow[row];
    const float izi = izrow[row];
    const unsigned int* brow = bits + (size_t)row * BITS_WORDS;

    v8f acc0 = {}, acc1 = {}, acc2 = {}, acc3 = {};
    for (int ks = 0; ks < CHUNK; ks += 32) {
        const int jb = jb0 + ks;
        const unsigned int word = brow[jb >> 5];
        // A fragment: pruned attention weights -> bf16 in contiguous k order (k = hi*16+e), as B
        v16bf a;
#pragma unroll
        for (int e = 0; e < 16; ++e) {
            int koff = hi * 16 + e;
            int lj = ks + koff;
            float ev = esi + led[lj];
            ev = ev > 0.f ? ev : ALPHA * ev;
            float p = __expf(ev - mi) * izi;
            float t = p - lth[lj];
            bool on = ((word >> koff) & 1u) && (t > 0.f);
            a[e] = (__bf16)(on ? t : 0.f);
        }
        // B fragments: Wh^T rows are contiguous in K -> single 32B load per tile
        const __bf16* bbase = WhBT + jb + hi * 16;
        v16bf b0 = *(const v16bf*)(bbase + (size_t)(0 * 16 + lrow) * WHT_STRIDE);
        v16bf b1 = *(const v16bf*)(bbase + (size_t)(1 * 16 + lrow) * WHT_STRIDE);
        v16bf b2 = *(const v16bf*)(bbase + (size_t)(2 * 16 + lrow) * WHT_STRIDE);
        v16bf b3 = *(const v16bf*)(bbase + (size_t)(3 * 16 + lrow) * WHT_STRIDE);
        acc0 = __builtin_amdgcn_wmma_f32_16x16x32_bf16(false, a, false, b0, (short)0, acc0, false, false);
        acc1 = __builtin_amdgcn_wmma_f32_16x16x32_bf16(false, a, false, b1, (short)0, acc1, false, false);
        acc2 = __builtin_amdgcn_wmma_f32_16x16x32_bf16(false, a, false, b2, (short)0, acc2, false, false);
        acc3 = __builtin_amdgcn_wmma_f32_16x16x32_bf16(false, a, false, b3, (short)0, acc3, false, false);
    }
    float* pbase = part + ((size_t)ch * N_NODES + i0) * OUT_F;
#pragma unroll
    for (int v = 0; v < 8; ++v) {
        int irow = hi * 8 + v;
        pbase[(size_t)irow * OUT_F + 0 * 16 + lrow] = acc0[v];
        pbase[(size_t)irow * OUT_F + 1 * 16 + lrow] = acc1[v];
        pbase[(size_t)irow * OUT_F + 2 * 16 + lrow] = acc2[v];
        pbase[(size_t)irow * OUT_F + 3 * 16 + lrow] = acc3[v];
    }
}

// ---------- kernel 4: reduce chunk partials + ELU ----------
__global__ void k_reduce(const float* __restrict__ part, float* __restrict__ out) {
    size_t idx = (size_t)blockIdx.x * blockDim.x + threadIdx.x;
    if (idx >= (size_t)N_NODES * OUT_F) return;
    float s = 0.f;
#pragma unroll
    for (int c = 0; c < NCH; ++c) s += part[(size_t)c * N_NODES * OUT_F + idx];
    out[idx] = s > 0.f ? s : (__expf(s) - 1.f);
}

extern "C" void kernel_launch(void* const* d_in, const int* in_sizes, int n_in,
                              void* d_out, int out_size, void* d_ws, size_t ws_size,
                              hipStream_t stream) {
    const float* h     = (const float*)d_in[0];
    const float* W     = (const float*)d_in[1];
    const float* a_src = (const float*)d_in[2];
    const float* a_dst = (const float*)d_in[3];
    const float* hub   = (const float*)d_in[4];
    const int*   adj   = (const int*)d_in[5];

    char* ws = (char*)d_ws;
    __bf16* hB    = (__bf16*)(ws + HB_OFF);
    __bf16* WT    = (__bf16*)(ws + WT_OFF);
    __bf16* WhBT  = (__bf16*)(ws + WHT_OFF);
    float*  es    = (float*)(ws + ES_OFF);
    float*  ed    = (float*)(ws + ED_OFF);
    float*  mrow  = (float*)(ws + M_OFF);
    float*  izrow = (float*)(ws + IZ_OFF);
    unsigned int* bits = (unsigned int*)(ws + BITS_OFF);
    float*  part  = (float*)(ws + PART_OFF);

    k_cvt_h<<<((size_t)N_NODES * IN_F + 255) / 256, 256, 0, stream>>>(h, hB);
    k_cvt_w<<<(IN_F * OUT_F + 255) / 256, 256, 0, stream>>>(W, WT, WhBT);
    k_wh<<<N_NODES / 16, 128, 0, stream>>>(hB, WT, a_src, a_dst, WhBT, es, ed);
    k_stats<<<N_NODES, 256, 0, stream>>>(adj, ed, es, bits, mrow, izrow);
    k_aggr<<<dim3(N_NODES / 16, NCH), 32, 0, stream>>>(WhBT, ed, es, hub, bits,
                                                       mrow, izrow, part);
    k_reduce<<<((size_t)N_NODES * OUT_F + 255) / 256, 256, 0, stream>>>(part, (float*)d_out);
}
